// ProtoNet_51754355917524
// MI455X (gfx1250) — compile-verified, matrix operand order checked by reading, not run on hardware
//
#include <hip/hip_runtime.h>
#include <hip/hip_bf16.h>

typedef __attribute__((ext_vector_type(16))) __bf16 v16bf;
typedef __attribute__((ext_vector_type(8)))  __bf16 v8bf;
typedef __attribute__((ext_vector_type(4)))  __bf16 v4bf;
typedef __attribute__((ext_vector_type(8)))  float  v8f;

#define N_WORDS 131072
#define N_TAGS  512
#define BDIM    768
#define MDIM    512

#define ROWS        64    // word/tag rows per block
#define MT          4     // M-tiles per wave (ROWS/16)
#define NT          4     // N-tiles per wave (8 waves * 4 * 16 = 512 cols)
#define BLOCK       256   // 8 waves
#define METB_STRIDE 520   // bf16 metric tile row stride (halfs): 1040B rows -> 4-bank skew
#define CHK_STRIDE  40    // bf16 K-chunk staging row stride (halfs): 80B rows

// Fragment layout (ISA 7.12.2, 16x16x32 bf16):
//  A: lane lp (0-15) holds row m=lp; elems 0-7 -> K=kbase+half*8+i,
//     elems 8-15 -> K=kbase+16+half*8+(i-8); both runs contiguous (16B each).
//  B mirrored with N = lp.  C/D: row = 8*half + j, col = lp.
__device__ __forceinline__ v16bf frag_from_bf16(const __bf16* p) {
  const v8bf lo = *(const v8bf*)(p);
  const v8bf hi = *(const v8bf*)(p + 16);
  return __builtin_shufflevector(lo, hi, 0,1,2,3,4,5,6,7,8,9,10,11,12,13,14,15);
}
__device__ __forceinline__ v16bf a_frag_lds(const __bf16* sm, int stride, int m,
                                            int kbase, int half) {
  return frag_from_bf16(sm + m * stride + kbase + half * 8);
}
__device__ __forceinline__ v16bf b_frag_glb(const __bf16* __restrict__ W, int ldw, int n,
                                            int kbase, int half) {
  return frag_from_bf16(W + (size_t)n * ldw + kbase + half * 8);
}

// butterfly sum/max over lanes 0-15 (and 16-31 independently): masks < 16
__device__ __forceinline__ float xor16_sum(float v) {
#pragma unroll
  for (int m = 8; m >= 1; m >>= 1) v += __shfl_xor(v, m, 32);
  return v;
}
__device__ __forceinline__ float xor16_max(float v) {
#pragma unroll
  for (int m = 8; m >= 1; m >>= 1) v = fmaxf(v, __shfl_xor(v, m, 32));
  return v;
}

// -------- prep: f32 weights -> bf16 (run once; L2-resident afterwards) --------
__global__ void __launch_bounds__(256)
convert_f32_to_bf16_kernel(const float* __restrict__ src, __bf16* __restrict__ dst, int n4) {
  const int i = blockIdx.x * 256 + threadIdx.x;
  if (i < n4) {
    const float4 v = *(const float4*)(src + (size_t)i * 4);
    v4bf o;
    o[0] = (__bf16)v.x; o[1] = (__bf16)v.y; o[2] = (__bf16)v.z; o[3] = (__bf16)v.w;
    *(v4bf*)(dst + (size_t)i * 4) = o;
  }
}

// -------- metric tile [ROWS x MDIM] = embed[r0..r0+ROWS) @ Wb^T + bias --------
// 8 waves x (MT M-tiles x NT N-tiles). Leaves bf16 tile + per-row sum-of-squares
// in LDS (sm_sq[ROWS]) and ends on a barrier.
__device__ __forceinline__ void project_tile(const float* __restrict__ embed,
                                             const __bf16* __restrict__ Wb,
                                             const float* __restrict__ bias,
                                             int r0, __bf16* sm_bf16, __bf16* sm_chunk,
                                             float* sm_red, float* sm_sq) {
  const int tid  = threadIdx.x;
  const int wave = tid >> 5;
  const int lane = tid & 31;
  const int half = lane >> 4;
  const int lp   = lane & 15;

  v8f acc[MT][NT];
#pragma unroll
  for (int nt = 0; nt < NT; ++nt) {
    const float bv = bias[(wave * NT + nt) * 16 + lp];
#pragma unroll
    for (int mt = 0; mt < MT; ++mt)
#pragma unroll
      for (int j = 0; j < 8; ++j) acc[mt][nt][j] = bv;   // fold bias into C
  }

  for (int kb = 0; kb < BDIM; kb += 32) {
    __syncthreads();
#pragma unroll
    for (int p = 0; p < (ROWS * 32 / 4) / BLOCK; ++p) {  // stage ROWSx32 chunk as bf16
      const int idx = p * BLOCK + tid;
      const int row = idx >> 3, koff = (idx & 7) << 2;
      const float4 v = *(const float4*)(embed + (size_t)(r0 + row) * BDIM + kb + koff);
      v4bf o;
      o[0] = (__bf16)v.x; o[1] = (__bf16)v.y; o[2] = (__bf16)v.z; o[3] = (__bf16)v.w;
      *(v4bf*)(sm_chunk + row * CHK_STRIDE + koff) = o;
    }
    __syncthreads();
    v16bf a[MT];
#pragma unroll
    for (int mt = 0; mt < MT; ++mt)
      a[mt] = a_frag_lds(sm_chunk, CHK_STRIDE, mt * 16 + lp, 0, half);
#pragma unroll
    for (int nt = 0; nt < NT; ++nt) {
      const int n = (wave * NT + nt) * 16 + lp;
      const v16bf b = b_frag_glb(Wb, BDIM, n, kb, half);   // reused by all M-tiles
#pragma unroll
      for (int mt = 0; mt < MT; ++mt)
        acc[mt][nt] = __builtin_amdgcn_wmma_f32_16x16x32_bf16(
            false, a[mt], false, b, (short)0, acc[mt][nt], false, false);
    }
  }

  // write bf16 tile + per-row sum-of-squares partials
#pragma unroll
  for (int mt = 0; mt < MT; ++mt) {
#pragma unroll
    for (int j = 0; j < 8; ++j) {
      const int mrow = mt * 16 + half * 8 + j;
      float ss = 0.f;
#pragma unroll
      for (int nt = 0; nt < NT; ++nt) {
        const int n = (wave * NT + nt) * 16 + lp;
        const float v = acc[mt][nt][j];
        sm_bf16[mrow * METB_STRIDE + n] = (__bf16)v;
        ss += v * v;
      }
      ss = xor16_sum(ss);
      if (lp == 0) sm_red[mrow * 8 + wave] = ss;
    }
  }
  __syncthreads();
  if (tid < ROWS) {
    float s = 0.f;
#pragma unroll
    for (int w = 0; w < 8; ++w) s += sm_red[tid * 8 + w];
    sm_sq[tid] = s;
  }
  __syncthreads();
}

__global__ void __launch_bounds__(BLOCK)
tags_project_kernel(const float* __restrict__ tags_embed,
                    const __bf16* __restrict__ tags_Wb,
                    const float* __restrict__ tags_b,
                    __bf16* __restrict__ tagsM, float* __restrict__ t_sq) {
  __shared__ __bf16 sm_bf16[ROWS * METB_STRIDE];
  __shared__ __bf16 sm_chunk[ROWS * CHK_STRIDE];
  __shared__ float  sm_red[ROWS * 8];
  __shared__ float  sm_sq[ROWS];

  const int r0  = blockIdx.x * ROWS;
  const int tid = threadIdx.x;
  project_tile(tags_embed, tags_Wb, tags_b, r0, sm_bf16, sm_chunk, sm_red, sm_sq);

  // coalesced 16B copies of the bf16 metric tile to global
  for (int it = 0; it < (ROWS * MDIM / 8) / BLOCK; ++it) {
    const int chunk = it * BLOCK + tid;            // chunks of 8 halfs
    const int r = chunk >> 6, c8 = (chunk & 63) << 3;
    const v8bf v = *(const v8bf*)(sm_bf16 + r * METB_STRIDE + c8);
    *(v8bf*)(tagsM + (size_t)(r0 + r) * MDIM + c8) = v;
  }
  if (tid < ROWS) t_sq[r0 + tid] = sm_sq[tid];
}

__global__ void __launch_bounds__(BLOCK)
words_fused_kernel(const float* __restrict__ words_embed,
                   const __bf16* __restrict__ words_Wb,
                   const float* __restrict__ words_b,
                   const __bf16* __restrict__ tagsM,
                   const float* __restrict__ t_sq,
                   float* __restrict__ out_log, float* __restrict__ out_pred) {
  __shared__ __bf16 sm_bf16[ROWS * METB_STRIDE];
  __shared__ __bf16 sm_chunk[ROWS * CHK_STRIDE];
  __shared__ float  sm_red[ROWS * 8];
  __shared__ float  sm_wsq[ROWS];
  __shared__ float  sm_rmax[ROWS];
  __shared__ float  sm_lse[ROWS];

  const int tid  = threadIdx.x;
  const int wave = tid >> 5;
  const int lane = tid & 31;
  const int half = lane >> 4;
  const int lp   = lane & 15;
  const int r0   = blockIdx.x * ROWS;

  // Stage 1: words_metric tile (bf16 in LDS) + row norms (sm_wsq)
  project_tile(words_embed, words_Wb, words_b, r0, sm_bf16, sm_chunk, sm_red, sm_wsq);

  // Stage 2: cross = metric_tile @ tags_metric^T (K = MDIM)
  v8f acc[MT][NT];
#pragma unroll
  for (int mt = 0; mt < MT; ++mt)
#pragma unroll
    for (int nt = 0; nt < NT; ++nt)
#pragma unroll
      for (int j = 0; j < 8; ++j) acc[mt][nt][j] = 0.f;

  for (int kb = 0; kb < MDIM; kb += 32) {
    v16bf a[MT];
#pragma unroll
    for (int mt = 0; mt < MT; ++mt)
      a[mt] = a_frag_lds(sm_bf16, METB_STRIDE, mt * 16 + lp, kb, half);
#pragma unroll
    for (int nt = 0; nt < NT; ++nt) {
      const int t = (wave * NT + nt) * 16 + lp;
      const v16bf b = b_frag_glb(tagsM, MDIM, t, kb, half);
#pragma unroll
      for (int mt = 0; mt < MT; ++mt)
        acc[mt][nt] = __builtin_amdgcn_wmma_f32_16x16x32_bf16(
            false, a[mt], false, b, (short)0, acc[mt][nt], false, false);
    }
  }

  // logits (in registers): l = 2*cross - w_sq[row] - t_sq[col]
  float tsq[NT];
#pragma unroll
  for (int nt = 0; nt < NT; ++nt) tsq[nt] = t_sq[(wave * NT + nt) * 16 + lp];
#pragma unroll
  for (int mt = 0; mt < MT; ++mt)
#pragma unroll
    for (int j = 0; j < 8; ++j) {
      const float wsq = sm_wsq[mt * 16 + half * 8 + j];
#pragma unroll
      for (int nt = 0; nt < NT; ++nt)
        acc[mt][nt][j] = 2.f * acc[mt][nt][j] - wsq - tsq[nt];
    }

  // row max across the block: shuffle butterfly + 8-wave LDS partials
  __syncthreads();                 // sm_red free for reuse
#pragma unroll
  for (int mt = 0; mt < MT; ++mt)
#pragma unroll
    for (int j = 0; j < 8; ++j) {
      float mx = acc[mt][0][j];
#pragma unroll
      for (int nt = 1; nt < NT; ++nt) mx = fmaxf(mx, acc[mt][nt][j]);
      mx = xor16_max(mx);
      if (lp == 0) sm_red[(mt * 16 + half * 8 + j) * 8 + wave] = mx;
    }
  __syncthreads();
  if (tid < ROWS) {
    float mx = sm_red[tid * 8];
#pragma unroll
    for (int w = 1; w < 8; ++w) mx = fmaxf(mx, sm_red[tid * 8 + w]);
    sm_rmax[tid] = mx;
  }
  __syncthreads();

  // row sum of exp
#pragma unroll
  for (int mt = 0; mt < MT; ++mt)
#pragma unroll
    for (int j = 0; j < 8; ++j) {
      const float rmx = sm_rmax[mt * 16 + half * 8 + j];
      float s = 0.f;
#pragma unroll
      for (int nt = 0; nt < NT; ++nt) s += __expf(acc[mt][nt][j] - rmx);
      s = xor16_sum(s);
      if (lp == 0) sm_red[(mt * 16 + half * 8 + j) * 8 + wave] = s;
    }
  __syncthreads();
  if (tid < ROWS) {
    float s = 0.f;
#pragma unroll
    for (int w = 0; w < 8; ++w) s += sm_red[tid * 8 + w];
    sm_lse[tid] = __logf(s);
  }
  __syncthreads();

  // outputs straight from registers (16-lane contiguous 64B segments)
#pragma unroll
  for (int mt = 0; mt < MT; ++mt)
#pragma unroll
    for (int j = 0; j < 8; ++j) {
      const int mrow = mt * 16 + half * 8 + j;
      const float rmx = sm_rmax[mrow], lse = sm_lse[mrow];
      const size_t rowbase = (size_t)(r0 + mrow) * N_TAGS;
#pragma unroll
      for (int nt = 0; nt < NT; ++nt) {
        const int n = (wave * NT + nt) * 16 + lp;
        const float l = acc[mt][nt][j] - rmx - lse;
        out_log[rowbase + n]  = l;
        out_pred[rowbase + n] = __expf(l);
      }
    }
}

extern "C" void kernel_launch(void* const* d_in, const int* in_sizes, int n_in,
                              void* d_out, int out_size, void* d_ws, size_t ws_size,
                              hipStream_t stream) {
  (void)in_sizes; (void)n_in; (void)out_size; (void)ws_size;
  const float* tags_embed  = (const float*)d_in[0];
  const float* words_embed = (const float*)d_in[1];
  const float* tags_W      = (const float*)d_in[2];
  const float* tags_b      = (const float*)d_in[3];
  const float* words_W     = (const float*)d_in[4];
  const float* words_b     = (const float*)d_in[5];

  float* out_log  = (float*)d_out;
  float* out_pred = out_log + (size_t)N_WORDS * N_TAGS;

  // workspace layout
  char* ws = (char*)d_ws;
  __bf16* tagsM    = (__bf16*)ws;                      // 512*512*2 = 512 KB
  float*  t_sq     = (float*)(ws + 524288);            // 2 KB
  __bf16* tags_Wb  = (__bf16*)(ws + 526336);           // 768 KB
  __bf16* words_Wb = (__bf16*)(ws + 526336 + 786432);  // 768 KB

  const int n4 = (N_TAGS * BDIM) / 4;
  convert_f32_to_bf16_kernel<<<(n4 + 255) / 256, 256, 0, stream>>>(tags_W,  tags_Wb,  n4);
  convert_f32_to_bf16_kernel<<<(n4 + 255) / 256, 256, 0, stream>>>(words_W, words_Wb, n4);

  tags_project_kernel<<<N_TAGS / ROWS, BLOCK, 0, stream>>>(
      tags_embed, tags_Wb, tags_b, tagsM, t_sq);
  words_fused_kernel<<<N_WORDS / ROWS, BLOCK, 0, stream>>>(
      words_embed, words_Wb, words_b, tagsM, t_sq, out_log, out_pred);
}
